// MicroExpertMoE_17343077941602
// MI455X (gfx1250) — matrix operand order repaired and checked by reading, not compiled, not run on hardware
//
#include <hip/hip_runtime.h>
#include <hip/hip_bf16.h>

// ---------------------------------------------------------------------------
// MicroExpertMoE forward for MI455X (gfx1250, wave32, WMMA).
//
// Pipeline (all on `stream`):
//   1) f32->bf16 conversion of x, W_gate, W_up, W_down into workspace
//   2) router kernel: f32 logits -> soft top-2 suppression -> softmax -> w[n,e]
//   3) fused gate/up kernel: hw[n, e*256+k] = w[n,e]*silu(x@Wg^T)*(x@Wu^T) (bf16)
//   4) down kernel: out[n,d] = hw[n,:] @ Wd_flat  ([8192,4096] x [4096,1024])
//
// GEMMs use v_wmma_f32_16x16x32_bf16 with f32 accumulation. Per K-step, all
// fragment loads are issued before the WMMA block (claused loads, staggered
// waits), and the A fragment for the next K-step is double-buffered.
//   A 16x32 (MxK): lane L (m=L%16, g=L/16): elems j<8 -> K=8g+j, j>=8 -> K=16+8g+(j-8)
//   B 32x16 (KxN): lane L (n=L%16, g=L/16): elems j<8 -> K=8g+j, j>=8 -> K=16+8g+(j-8), col n
//   C 16x16 f32:   lane L (n=L%16, g=L/16): c[v] = C[m=v+8g][n]
// ---------------------------------------------------------------------------

typedef __attribute__((ext_vector_type(16))) __bf16 v16bf;
typedef __attribute__((ext_vector_type(8)))  float  v8f;

#define N_TOK 8192
#define DIM   1024
#define NEXP  16
#define KOUT  256
#define EK    (NEXP * KOUT)   // 4096

struct Q2 { uint4 lo, hi; };  // 32 bytes == 16 bf16

static __device__ __forceinline__ v16bf load16(const __bf16* p0, const __bf16* p1) {
  Q2 r;
  r.lo = *reinterpret_cast<const uint4*>(p0);   // global_load_b128
  r.hi = *reinterpret_cast<const uint4*>(p1);   // global_load_b128
  return __builtin_bit_cast(v16bf, r);
}

// ---------------------------------------------------------------------------
// 1) elementwise f32 -> bf16
// ---------------------------------------------------------------------------
__global__ void f32_to_bf16_kernel(const float* __restrict__ in,
                                   __bf16* __restrict__ out, long n) {
  long i = (long)blockIdx.x * blockDim.x + threadIdx.x;
  long stride = (long)gridDim.x * blockDim.x;
  for (; i < n; i += stride) out[i] = (__bf16)in[i];
}

// ---------------------------------------------------------------------------
// 2) router: one block (256 threads) per token, full f32 precision.
//    logits[e] = (x[n,:] . Wr[e,:]) / temp; soft top-2 suppression; softmax.
// ---------------------------------------------------------------------------
__global__ void __launch_bounds__(256) router_kernel(
    const float* __restrict__ x, const float* __restrict__ Wr,
    const float* __restrict__ log_temp, float* __restrict__ wout) {
  __shared__ float xs[DIM];
  __shared__ float part[NEXP][16];
  __shared__ float logits[NEXP];
  const int n = blockIdx.x;
  const int t = threadIdx.x;

  for (int d = t; d < DIM; d += 256) xs[d] = x[(long)n * DIM + d];
  __syncthreads();

  const int e = t >> 4, l = t & 15;
  float acc = 0.f;
  const float* wr = Wr + (long)e * DIM;
  for (int d = l; d < DIM; d += 16) acc += xs[d] * wr[d];
  part[e][l] = acc;
  __syncthreads();

  if (t < NEXP) {
    float s = 0.f;
    #pragma unroll
    for (int i = 0; i < 16; i++) s += part[t][i];
    float lt = log_temp[0];
    float sp = (lt > 20.f) ? lt : log1pf(__expf(lt));   // softplus
    logits[t] = s / (sp + 0.1f);
  }
  __syncthreads();

  if (t == 0) {
    float m1 = -INFINITY, m2 = -INFINITY;
    #pragma unroll
    for (int i = 0; i < NEXP; i++) {
      float v = logits[i];
      if (v > m1) { m2 = m1; m1 = v; } else if (v > m2) { m2 = v; }
    }
    const float thr = m2;                    // 2nd-largest == TOP_K threshold
    float g[NEXP], mx = -INFINITY;
    #pragma unroll
    for (int i = 0; i < NEXP; i++) {
      float v = logits[i];
      float sup = v / (1.f + __expf(-10.f * (v - thr)));  // v * sigmoid(10*(v-thr))
      g[i] = sup;
      mx = fmaxf(mx, sup);
    }
    float ssum = 0.f;
    #pragma unroll
    for (int i = 0; i < NEXP; i++) { g[i] = __expf(g[i] - mx); ssum += g[i]; }
    const float inv = 1.f / ssum;
    #pragma unroll
    for (int i = 0; i < NEXP; i++) wout[(long)n * NEXP + i] = g[i] * inv;
  }
}

// ---------------------------------------------------------------------------
// 3) fused gate/up: each wave computes a 16(M tokens) x 64(N k_out) tile for
//    one expert; gate and up share the A fragment. All 8 B-fragment loads are
//    issued before the 8 WMMAs; next A fragment is double-buffered. Epilogue
//    folds in silu and the router weight, stores bf16 to hw[n, e*256 + k].
// ---------------------------------------------------------------------------
__global__ void __launch_bounds__(256) moe_gateup_kernel(
    const __bf16* __restrict__ xb, const __bf16* __restrict__ Wg,
    const __bf16* __restrict__ Wu, const float* __restrict__ wrt,
    __bf16* __restrict__ hw) {
  const int wid  = blockIdx.x * 8 + (threadIdx.x >> 5);
  const int lane = threadIdx.x & 31;
  const int tm  = wid >> 6;          // 512 M-tiles
  const int rem = wid & 63;
  const int e   = rem >> 2;          // 16 experts
  const int tn  = rem & 3;           // 4 N-tiles of 64
  const int m_base = tm * 16;
  const int kout_base = tn * 64;
  const int ln = lane & 15, lg = lane >> 4;

  const __bf16* arow = xb + (long)(m_base + ln) * DIM + 8 * lg;
  const __bf16* pg[4];
  const __bf16* pu[4];
  #pragma unroll
  for (int t = 0; t < 4; t++) {
    long rowoff = (long)(kout_base + 16 * t + ln) * DIM + 8 * lg;
    pg[t] = Wg + (long)e * KOUT * DIM + rowoff;
    pu[t] = Wu + (long)e * KOUT * DIM + rowoff;
  }

  v8f cg[4] = {}, cu[4] = {};
  v16bf a_nxt = load16(arow, arow + 16);

  for (int kb = 0; kb < DIM; kb += 32) {
    v16bf a = a_nxt;
    v16bf bg[4], bu[4];
    #pragma unroll
    for (int t = 0; t < 4; t++) {
      bg[t] = load16(pg[t] + kb, pg[t] + kb + 16);
      bu[t] = load16(pu[t] + kb, pu[t] + kb + 16);
    }
    const int kn = (kb + 32 < DIM) ? (kb + 32) : 0;   // uniform select, no branch
    a_nxt = load16(arow + kn, arow + kn + 16);
    __builtin_prefetch(arow + kb + 256, 0, 1);        // speculative, safe OOB
    #pragma unroll
    for (int t = 0; t < 4; t++) {
      cg[t] = __builtin_amdgcn_wmma_f32_16x16x32_bf16(false, a, false, bg[t],
                                                      (short)0, cg[t], false, false);
      cu[t] = __builtin_amdgcn_wmma_f32_16x16x32_bf16(false, a, false, bu[t],
                                                      (short)0, cu[t], false, false);
    }
  }

  float wr[8];
  #pragma unroll
  for (int v = 0; v < 8; v++)
    wr[v] = wrt[(long)(m_base + v + 8 * lg) * NEXP + e];

  #pragma unroll
  for (int t = 0; t < 4; t++) {
    #pragma unroll
    for (int v = 0; v < 8; v++) {
      float g = cg[t][v];
      float s = g / (1.f + __expf(-g));          // silu
      float h = s * cu[t][v] * wr[v];
      int row = m_base + v + 8 * lg;
      int col = e * KOUT + kout_base + 16 * t + ln;
      hw[(long)row * EK + col] = (__bf16)h;
    }
  }
}

// ---------------------------------------------------------------------------
// 4) down projection: out[8192,1024] = hw[8192,4096] @ WdT, where the flat
//    K index (e*256 + k) maps to Wd[e, d, k] (k contiguous -> clean B loads,
//    32 | 256 so every K-step stays inside one expert block).
// ---------------------------------------------------------------------------
__global__ void __launch_bounds__(256) moe_down_kernel(
    const __bf16* __restrict__ hw, const __bf16* __restrict__ Wd,
    float* __restrict__ out) {
  const int wid  = blockIdx.x * 8 + (threadIdx.x >> 5);
  const int lane = threadIdx.x & 31;
  const int tm = wid >> 4;           // 512 M-tiles
  const int tn = wid & 15;           // 16 N-tiles of 64 over D=1024
  const int m_base = tm * 16;
  const int d_base = tn * 64;
  const int ln = lane & 15, lg = lane >> 4;

  const __bf16* arow = hw + (long)(m_base + ln) * EK + 8 * lg;
  const __bf16* pb[4];
  #pragma unroll
  for (int t = 0; t < 4; t++)
    pb[t] = Wd + (long)(d_base + 16 * t + ln) * KOUT + 8 * lg;

  v8f c[4] = {};
  v16bf a_nxt = load16(arow, arow + 16);

  #pragma unroll 2
  for (int kb = 0; kb < EK; kb += 32) {
    v16bf a = a_nxt;
    // flat K (e*256+kin) -> Wd offset e*DIM*KOUT + kin, uniform per K-step
    const long boff = (long)(kb >> 8) * (DIM * KOUT) + (kb & 255);
    v16bf b[4];
    #pragma unroll
    for (int t = 0; t < 4; t++)
      b[t] = load16(pb[t] + boff, pb[t] + boff + 16);
    const int kn = (kb + 32 < EK) ? (kb + 32) : 0;    // uniform select, no branch
    a_nxt = load16(arow + kn, arow + kn + 16);
    __builtin_prefetch(arow + kb + 256, 0, 1);        // speculative, safe OOB
    #pragma unroll
    for (int t = 0; t < 4; t++)
      c[t] = __builtin_amdgcn_wmma_f32_16x16x32_bf16(false, a, false, b[t],
                                                     (short)0, c[t], false, false);
  }

  #pragma unroll
  for (int t = 0; t < 4; t++)
    #pragma unroll
    for (int v = 0; v < 8; v++)
      out[(long)(m_base + v + 8 * lg) * DIM + d_base + 16 * t + ln] = c[t][v];
}

// ---------------------------------------------------------------------------
extern "C" void kernel_launch(void* const* d_in, const int* in_sizes, int n_in,
                              void* d_out, int out_size, void* d_ws, size_t ws_size,
                              hipStream_t stream) {
  const float* x  = (const float*)d_in[0];
  const float* Wu = (const float*)d_in[1];
  const float* Wg = (const float*)d_in[2];
  const float* Wd = (const float*)d_in[3];
  const float* Wr = (const float*)d_in[4];
  const float* lt = (const float*)d_in[5];
  float* out = (float*)d_out;

  char* ws = (char*)d_ws;
  __bf16* xb  = (__bf16*)ws; ws += (size_t)N_TOK * DIM * 2;        // 16 MB
  __bf16* Wgb = (__bf16*)ws; ws += (size_t)NEXP * KOUT * DIM * 2;  //  8 MB
  __bf16* Wub = (__bf16*)ws; ws += (size_t)NEXP * KOUT * DIM * 2;  //  8 MB
  __bf16* Wdb = (__bf16*)ws; ws += (size_t)NEXP * DIM * KOUT * 2;  //  8 MB
  float*  wrt = (float*)ws;  ws += (size_t)N_TOK * NEXP * 4;       // 0.5 MB
  __bf16* hwb = (__bf16*)ws;                                       // 64 MB

  f32_to_bf16_kernel<<<2048, 256, 0, stream>>>(x,  xb,  (long)N_TOK * DIM);
  f32_to_bf16_kernel<<<1024, 256, 0, stream>>>(Wg, Wgb, (long)NEXP * KOUT * DIM);
  f32_to_bf16_kernel<<<1024, 256, 0, stream>>>(Wu, Wub, (long)NEXP * KOUT * DIM);
  f32_to_bf16_kernel<<<1024, 256, 0, stream>>>(Wd, Wdb, (long)NEXP * DIM * KOUT);

  router_kernel<<<N_TOK, 256, 0, stream>>>(x, Wr, lt, wrt);

  // 512 M-tiles * 16 experts * 4 N-tiles = 32768 waves / 8 per block
  moe_gateup_kernel<<<4096, 256, 0, stream>>>(xb, Wgb, Wub, wrt, hwb);
  // 512 M-tiles * 16 N-tiles = 8192 waves / 8 per block
  moe_down_kernel<<<1024, 256, 0, stream>>>(hwb, Wdb, out);
}
